// NamedEntityRecognitionModel_40243843564128
// MI455X (gfx1250) — matrix operand order repaired and checked by reading, not compiled, not run on hardware
//
#include <hip/hip_runtime.h>
#include <hip/hip_bf16.h>

typedef __bf16 bf16_t;
typedef __attribute__((ext_vector_type(16))) __bf16 v16bf;
typedef __attribute__((ext_vector_type(8)))  __bf16 v8bf;
typedef __attribute__((ext_vector_type(8)))  float  v8f;

constexpr int S_ = 512, B_ = 32, D_ = 512, H_ = 512, V_ = 64;
constexpr int G4_ = 4 * H_;   // 2048 gate rows
constexpr int NWG_ = 32;      // workgroups per LSTM cell in the scan

// ---------------- utility kernels ----------------
__global__ void k_zero_i32(int* p, int n) {
  int i = blockIdx.x * 256 + threadIdx.x;
  if (i < n) p[i] = 0;
}

__global__ void k_f32_to_bf16(const float* __restrict__ s, bf16_t* __restrict__ d, int n) {
  int i = blockIdx.x * 256 + threadIdx.x;
  if (i < n) d[i] = (bf16_t)s[i];
}

// embedding gather for batch 0 only, emitted directly as bf16
__global__ void k_embed_bf16(const int* __restrict__ src, const float* __restrict__ emb,
                             bf16_t* __restrict__ x, int n) {
  int i = blockIdx.x * 256 + threadIdx.x;
  if (i >= n) return;
  int s = i / D_;
  int d = i - s * D_;
  int tok = src[s * B_];  // batch 0
  x[i] = (bf16_t)emb[(size_t)tok * D_ + d];
}

// ---------------- WMMA GEMM: C[M,N] = A[M,K](bf16) * W[N,K]^T(bf16) + bias ----------------
// One wave per block; each wave computes a 16x64 tile (4 N-subtiles) accumulating K in steps of 32.
union BfFrag { v16bf v; uint4 u[2]; };

__global__ void k_wmma_gemm(const bf16_t* __restrict__ A, int lda,
                            const bf16_t* __restrict__ W, int ldw,
                            float* __restrict__ C, int ldc,
                            const float* __restrict__ bias0,
                            const float* __restrict__ bias1,
                            int K) {
  const int m0   = blockIdx.x * 16;
  const int n0   = blockIdx.y * 64;
  const int lane = threadIdx.x & 31;
  const int row  = lane & 15;
  const int hi   = lane >> 4;

  v8f acc[4] = {v8f{}, v8f{}, v8f{}, v8f{}};

  // A fragment base: lane<16 holds M=row, K offsets {0..15}; lane>=16 holds K {16..31}
  const bf16_t* Ap = A + (size_t)(m0 + row) * lda + hi * 16;
  // B fragment bases: lane gives column n0+nt*16+row, hi half selects K-half (16 contiguous)
  const bf16_t* Wp0 = W + (size_t)(n0 +  0 + row) * ldw + hi * 16;
  const bf16_t* Wp1 = W + (size_t)(n0 + 16 + row) * ldw + hi * 16;
  const bf16_t* Wp2 = W + (size_t)(n0 + 32 + row) * ldw + hi * 16;
  const bf16_t* Wp3 = W + (size_t)(n0 + 48 + row) * ldw + hi * 16;

  for (int k = 0; k < K; k += 32) {
    BfFrag a;
    a.u[0] = *reinterpret_cast<const uint4*>(Ap + k);        // K = k+hi*16 .. +7
    a.u[1] = *reinterpret_cast<const uint4*>(Ap + k + 8);    // K = k+hi*16+8 .. +15

    BfFrag b0, b1, b2, b3;
    b0.u[0] = *reinterpret_cast<const uint4*>(Wp0 + k);
    b0.u[1] = *reinterpret_cast<const uint4*>(Wp0 + k + 8);
    b1.u[0] = *reinterpret_cast<const uint4*>(Wp1 + k);
    b1.u[1] = *reinterpret_cast<const uint4*>(Wp1 + k + 8);
    b2.u[0] = *reinterpret_cast<const uint4*>(Wp2 + k);
    b2.u[1] = *reinterpret_cast<const uint4*>(Wp2 + k + 8);
    b3.u[0] = *reinterpret_cast<const uint4*>(Wp3 + k);
    b3.u[1] = *reinterpret_cast<const uint4*>(Wp3 + k + 8);

    acc[0] = __builtin_amdgcn_wmma_f32_16x16x32_bf16(false, a.v, false, b0.v, (short)0, acc[0], false, false);
    acc[1] = __builtin_amdgcn_wmma_f32_16x16x32_bf16(false, a.v, false, b1.v, (short)0, acc[1], false, false);
    acc[2] = __builtin_amdgcn_wmma_f32_16x16x32_bf16(false, a.v, false, b2.v, (short)0, acc[2], false, false);
    acc[3] = __builtin_amdgcn_wmma_f32_16x16x32_bf16(false, a.v, false, b3.v, (short)0, acc[3], false, false);
  }

  // C/D layout: VGPR r -> (M = m0 + r + 8*hi, N = n0 + nt*16 + (lane&15))
  #pragma unroll
  for (int nt = 0; nt < 4; ++nt) {
    int n = n0 + nt * 16 + row;
    float bsum = bias0 ? bias0[n] : 0.0f;
    if (bias1) bsum += bias1[n];
    #pragma unroll
    for (int r = 0; r < 8; ++r) {
      int m = m0 + r + hi * 8;
      C[(size_t)m * ldc + n] = acc[nt][r] + bsum;
    }
  }
}

// ---------------- persistent LSTM scan (one layer, both directions) ----------------
// grid = (NWG_, 2): blockIdx.y = cell (0=fwd, 1=bwd), 256 threads.
// Each WG owns 16 hidden indices (j0..j0+15) -> 64 gate rows; cell state lives in LDS.
// Whh is pre-converted to bf16 (halves L2 stream per step). Per-step cross-WG sequencing
// via global arrive counters; a cluster barrier (NOP outside clusters) is issued as the
// hardware fast path.
union W8 { uint4 u; v8bf b; };

__global__ void k_lstm_scan(const float*  __restrict__ Gin,    // [2][S][4H] (input proj + biases)
                            const bf16_t* __restrict__ WhhBf,  // [2][4H][H] bf16
                            float* hcur,                        // [2][H]
                            bf16_t* __restrict__ XoutBf,       // [S][2H] bf16
                            int* bar)                           // [2][S]
{
  const int wg   = blockIdx.x;
  const int cell = blockIdx.y;
  const int tid  = threadIdx.x;
  const int r    = tid >> 2;          // 0..63 local gate row
  const int q    = tid & 3;           // K quarter
  const int j0   = wg * 16;
  const int gate = r >> 4;            // 0=i 1=f 2=g 3=o
  const int g    = gate * H_ + j0 + (r & 15);   // global gate row

  __shared__ float h_sh[H_];
  __shared__ float gates_sh[64];
  __shared__ float c_sh[16];
  if (tid < 16) c_sh[tid] = 0.0f;

  const uint4* wp = reinterpret_cast<const uint4*>(
      WhhBf + ((size_t)cell * G4_ + g) * H_ + q * 128);   // 128 bf16 = 16 x uint4
  const float* GinCell = Gin + (size_t)cell * S_ * G4_;
  volatile const float* hv = hcur;

  for (int step = 0; step < S_; ++step) {
    const int tt = (cell == 1) ? (S_ - 1 - step) : step;

    // load previous h (all WGs' contributions) into LDS
    for (int i = tid; i < H_; i += 256)
      h_sh[i] = (step == 0) ? 0.0f : hv[cell * H_ + i];
    // prefetch next timestep's gate pre-activation (global_prefetch_b8)
    if (step + 1 < S_) {
      const int tn = (cell == 1) ? (S_ - 2 - step) : (step + 1);
      __builtin_prefetch(GinCell + (size_t)tn * G4_ + g, 0, 3);
    }
    __syncthreads();

    // partial dot: 128 bf16 weights * 128 f32 h elements
    const float* hh = h_sh + q * 128;
    float s = 0.0f;
    #pragma unroll 4
    for (int i = 0; i < 16; ++i) {
      W8 w; w.u = wp[i];
      const float* hp = hh + i * 8;
      s += (float)w.b[0] * hp[0] + (float)w.b[1] * hp[1] +
           (float)w.b[2] * hp[2] + (float)w.b[3] * hp[3] +
           (float)w.b[4] * hp[4] + (float)w.b[5] * hp[5] +
           (float)w.b[6] * hp[6] + (float)w.b[7] * hp[7];
    }
    s += __shfl_xor(s, 1);
    s += __shfl_xor(s, 2);
    if (q == 0)
      gates_sh[r] = GinCell[(size_t)tt * G4_ + g] + s;
    __syncthreads();

    // gate math + state update for our 16 hidden units
    if (tid < 16) {
      const int jl = tid;
      float gi = gates_sh[jl];
      float gf = gates_sh[16 + jl];
      float gg = gates_sh[32 + jl];
      float go = gates_sh[48 + jl];
      float si = 1.0f / (1.0f + __expf(-gi));
      float sf = 1.0f / (1.0f + __expf(-gf));
      float so = 1.0f / (1.0f + __expf(-go));
      float c  = sf * c_sh[jl] + si * tanhf(gg);
      float h  = so * tanhf(c);
      c_sh[jl] = c;
      int j = j0 + jl;
      hcur[cell * H_ + j] = h;
      XoutBf[(size_t)tt * (2 * H_) + cell * H_ + j] = (bf16_t)h;
    }
    __threadfence();
    __syncthreads();

    // hardware cluster barrier fast path (s_barrier_signal/-wait -3; NOP if not clustered)
    __builtin_amdgcn_s_cluster_barrier();

    // arrive + spin: all NWG_ workgroups of this cell finish step before anyone starts step+1
    if (tid == 0) {
      int bi = cell * S_ + step;
      atomicAdd(&bar[bi], 1);
      while (((volatile int*)bar)[bi] < NWG_) __builtin_amdgcn_s_sleep(2);
    }
    __syncthreads();
    __threadfence();
  }
}

// ---------------- Viterbi (V=64): one block, delta in LDS ----------------
__global__ void k_viterbi(const float* __restrict__ logits,  // [S][V]
                          const float* __restrict__ trans,   // [V][V]
                          int* __restrict__ bps,             // [S-1][V]
                          float* __restrict__ out)           // [S]
{
  __shared__ float delta[V_];
  const int v = threadIdx.x;
  delta[v] = logits[v];
  __syncthreads();

  for (int t = 1; t < S_; ++t) {
    float best = -3.0e38f; int bu = 0;
    #pragma unroll 8
    for (int u = 0; u < V_; ++u) {
      float sc = delta[u] + trans[u * V_ + v];
      if (sc > best) { best = sc; bu = u; }
    }
    best += logits[t * V_ + v];
    bps[(t - 1) * V_ + v] = bu;
    __syncthreads();
    delta[v] = best;
    __syncthreads();
  }

  if (v == 0) {
    int last = 0; float b = delta[0];
    for (int u = 1; u < V_; ++u) if (delta[u] > b) { b = delta[u]; last = u; }
    out[S_ - 1] = (float)last;
    for (int k = S_ - 2; k >= 0; --k) {
      last = bps[k * V_ + last];
      out[k] = (float)last;
    }
  }
}

// ---------------- host orchestration ----------------
extern "C" void kernel_launch(void* const* d_in, const int* in_sizes, int n_in,
                              void* d_out, int out_size, void* d_ws, size_t ws_size,
                              hipStream_t stream) {
  const int*   source = (const int*)  d_in[0];
  const float* emb    = (const float*)d_in[2];
  const float* Wih0   = (const float*)d_in[3];
  const float* Whh0   = (const float*)d_in[4];
  const float* bih0   = (const float*)d_in[5];
  const float* bhh0   = (const float*)d_in[6];
  const float* Wih1   = (const float*)d_in[7];
  const float* Whh1   = (const float*)d_in[8];
  const float* bih1   = (const float*)d_in[9];
  const float* bhh1   = (const float*)d_in[10];
  const float* Wout   = (const float*)d_in[11];
  const float* bout   = (const float*)d_in[12];
  const float* trans  = (const float*)d_in[13];
  float* out = (float*)d_out;

  // workspace carve-up
  size_t o = 0;
  auto alloc = [&](size_t bytes) { size_t r = o; o = (o + bytes + 255) & ~(size_t)255; return r; };
  char* ws = (char*)d_ws;
  bf16_t* xbf     = (bf16_t*)(ws + alloc((size_t)S_ * D_ * 2));
  bf16_t* Wih0bf  = (bf16_t*)(ws + alloc((size_t)2 * G4_ * D_ * 2));
  bf16_t* Wih1bf  = (bf16_t*)(ws + alloc((size_t)2 * G4_ * (2 * H_) * 2));
  bf16_t* Whh0bf  = (bf16_t*)(ws + alloc((size_t)2 * G4_ * H_ * 2));
  bf16_t* Whh1bf  = (bf16_t*)(ws + alloc((size_t)2 * G4_ * H_ * 2));
  bf16_t* Woutbf  = (bf16_t*)(ws + alloc((size_t)V_ * (2 * H_) * 2));
  float*  Gin0    = (float*) (ws + alloc((size_t)2 * S_ * G4_ * 4));
  float*  Gin1    = (float*) (ws + alloc((size_t)2 * S_ * G4_ * 4));
  bf16_t* X1bf    = (bf16_t*)(ws + alloc((size_t)S_ * 2 * H_ * 2));
  bf16_t* X2bf    = (bf16_t*)(ws + alloc((size_t)S_ * 2 * H_ * 2));
  float*  hcur    = (float*) (ws + alloc((size_t)2 * H_ * 4));
  int*    bars    = (int*)   (ws + alloc((size_t)2 * 2 * S_ * 4));   // [layer][cell][step]
  float*  logits  = (float*) (ws + alloc((size_t)S_ * V_ * 4));
  int*    bps     = (int*)   (ws + alloc((size_t)(S_ - 1) * V_ * 4));
  (void)ws_size; (void)in_sizes; (void)n_in; (void)out_size;

  // 0) reset per-step barriers (required on every graph replay)
  {
    int n = 2 * 2 * S_;
    k_zero_i32<<<(n + 255) / 256, 256, 0, stream>>>(bars, n);
  }
  // 1) embedding gather (batch 0) -> bf16
  {
    int n = S_ * D_;
    k_embed_bf16<<<(n + 255) / 256, 256, 0, stream>>>(source, emb, xbf, n);
  }
  // 2) weight conversions f32 -> bf16 (one pass; all stay L2-resident)
  {
    int n0 = 2 * G4_ * D_;
    k_f32_to_bf16<<<(n0 + 255) / 256, 256, 0, stream>>>(Wih0, Wih0bf, n0);
    int n1 = 2 * G4_ * (2 * H_);
    k_f32_to_bf16<<<(n1 + 255) / 256, 256, 0, stream>>>(Wih1, Wih1bf, n1);
    int n2 = 2 * G4_ * H_;
    k_f32_to_bf16<<<(n2 + 255) / 256, 256, 0, stream>>>(Whh0, Whh0bf, n2);
    k_f32_to_bf16<<<(n2 + 255) / 256, 256, 0, stream>>>(Whh1, Whh1bf, n2);
    int n3 = V_ * (2 * H_);
    k_f32_to_bf16<<<(n3 + 255) / 256, 256, 0, stream>>>(Wout, Woutbf, n3);
  }
  // 3) layer-0 input projections: Gin0[cell] = x @ Wih0[cell]^T + bih0 + bhh0   (WMMA)
  for (int cell = 0; cell < 2; ++cell) {
    dim3 grid(S_ / 16, G4_ / 64);
    k_wmma_gemm<<<grid, 32, 0, stream>>>(
        xbf, D_,
        Wih0bf + (size_t)cell * G4_ * D_, D_,
        Gin0 + (size_t)cell * S_ * G4_, G4_,
        bih0 + cell * G4_, bhh0 + cell * G4_, D_);
  }
  // 4) layer-0 recurrent scan (fwd + bwd in parallel)
  {
    dim3 grid(NWG_, 2);
    k_lstm_scan<<<grid, 256, 0, stream>>>(Gin0, Whh0bf, hcur, X1bf, bars + 0);
  }
  // 5) layer-1 input projections: Gin1[cell] = X1 @ Wih1[cell]^T + biases   (WMMA, K=1024)
  for (int cell = 0; cell < 2; ++cell) {
    dim3 grid(S_ / 16, G4_ / 64);
    k_wmma_gemm<<<grid, 32, 0, stream>>>(
        X1bf, 2 * H_,
        Wih1bf + (size_t)cell * G4_ * (2 * H_), 2 * H_,
        Gin1 + (size_t)cell * S_ * G4_, G4_,
        bih1 + cell * G4_, bhh1 + cell * G4_, 2 * H_);
  }
  // 6) layer-1 recurrent scan
  {
    dim3 grid(NWG_, 2);
    k_lstm_scan<<<grid, 256, 0, stream>>>(Gin1, Whh1bf, hcur, X2bf, bars + 2 * S_);
  }
  // 7) logits = X2 @ Wout^T + bout   (WMMA, M=512 N=64 K=1024)
  {
    dim3 grid(S_ / 16, V_ / 64);
    k_wmma_gemm<<<grid, 32, 0, stream>>>(
        X2bf, 2 * H_,
        Woutbf, 2 * H_,
        logits, V_,
        bout, nullptr, 2 * H_);
  }
  // 8) Viterbi decode -> path (as floats per harness output convention)
  k_viterbi<<<1, V_, 0, stream>>>(logits, trans, bps, out);
}
